// GCN_LSTM_BI_Attention_37890201486073
// MI455X (gfx1250) — matrix operand order repaired and checked by reading, not compiled, not run on hardware
//
#include <hip/hip_runtime.h>
#include <hip/hip_bf16.h>

// Problem constants (from reference)
#define GN   10000
#define GE   320000
#define GINC 64
#define GH   128
#define GHP  129   // padded LDS row stride (bank-conflict-free)
#define G4H  512
#define GP   12
#define GEPS 1e-5f

typedef __attribute__((ext_vector_type(16))) __bf16 v16bf;
typedef __attribute__((ext_vector_type(8)))  float  v8f;

__device__ __forceinline__ float sigf(float x) { return 1.f / (1.f + __expf(-x)); }

__device__ __forceinline__ void cvt4(v16bf& v, int base, float4 f) {
  v[base + 0] = (__bf16)f.x;
  v[base + 1] = (__bf16)f.y;
  v[base + 2] = (__bf16)f.z;
  v[base + 3] = (__bf16)f.w;
}

// ---------------------------------------------------------------------------
// Small transpose: out[c*rows + r] = in[r*cols + c]  (weights only; tiny)
// ---------------------------------------------------------------------------
__global__ void transpose_mat(const float* __restrict__ in,
                              float* __restrict__ out, int rows, int cols) {
  const int total  = rows * cols;
  const int stride = gridDim.x * blockDim.x;
  for (int i = blockIdx.x * blockDim.x + threadIdx.x; i < total; i += stride) {
    const int r = i / cols;
    const int c = i % cols;
    out[c * rows + r] = in[i];
  }
}

// ---------------------------------------------------------------------------
// WMMA bf16 GEMM:  C[M,Nc] = A[M,K] @ B^T (+bias1+bias2 per column)
//   B stored [Nc,K] row-major (elem (k,n) at B[n*K+k]) -> all loads contiguous
// Grid: (Nc/64, ceil(M/64)); block: 128 threads (4 waves).
// Each wave: 16x64 of C = one A fragment x 4 B fragments x 4 accumulators
// per K-step (4x A-reuse). All fragment loads are 16B-aligned float4 (b128).
// Requires M%16==0, K%32==0, Nc%64==0.
// ---------------------------------------------------------------------------
__global__ void gemm_wmma_bf16(const float* __restrict__ A,
                               const float* __restrict__ B,
                               float* __restrict__ C,
                               int M, int K, int Nc,
                               const float* __restrict__ bias1,
                               const float* __restrict__ bias2) {
  const int lane  = threadIdx.x & 31;
  const int wave  = threadIdx.x >> 5;
  const int m0    = blockIdx.y * 64 + wave * 16;
  const int n0    = blockIdx.x * 64;
  if (m0 >= M) return;

  const int lmod  = lane & 15;
  const int lhalf = lane >> 4;

  v8f acc[4] = {};

  // A fragment: lane holds row (m0+lmod), K = klo..klo+15
  const int arow = m0 + lmod;
  const int klo  = lhalf * 16;
  // B fragment: lane holds column (ntile+lmod), K = bkof..bkof+15
  const int bkof = lhalf * 16;

  for (int kb = 0; kb < K; kb += 32) {
    v16bf af;
    const float* ap = A + (long long)arow * K + kb + klo;  // 16B-aligned
    cvt4(af, 0,  *(const float4*)(ap + 0));
    cvt4(af, 4,  *(const float4*)(ap + 4));
    cvt4(af, 8,  *(const float4*)(ap + 8));
    cvt4(af, 12, *(const float4*)(ap + 12));
#pragma unroll
    for (int s = 0; s < 4; ++s) {
      const int bcol = n0 + s * 16 + lmod;
      const float* bp = B + (long long)bcol * K + kb + bkof;  // 16B-aligned
      v16bf bfv;
      cvt4(bfv, 0,  *(const float4*)(bp + 0));
      cvt4(bfv, 4,  *(const float4*)(bp + 4));
      cvt4(bfv, 8,  *(const float4*)(bp + 8));
      cvt4(bfv, 12, *(const float4*)(bp + 12));
      acc[s] = __builtin_amdgcn_wmma_f32_16x16x32_bf16(
          false, af, false, bfv, (short)0, acc[s], false, false);
    }
  }

#pragma unroll
  for (int s = 0; s < 4; ++s) {
    const int bcol = n0 + s * 16 + lmod;
    float bias = 0.f;
    if (bias1) bias += bias1[bcol];
    if (bias2) bias += bias2[bcol];
#pragma unroll
    for (int r = 0; r < 8; ++r) {
      const int row = m0 + r + lhalf * 8;  // C/D layout: VGPR r -> M=r (+8 hi)
      C[(long long)row * Nc + bcol] = acc[s][r] + bias;
    }
  }
}

// ---------------------------------------------------------------------------
// agg[i][c] = bias[c]   (init before scatter; re-run every call -> graph-safe)
// ---------------------------------------------------------------------------
__global__ void fill_bias(float* __restrict__ out, const float* __restrict__ b,
                          int n, int h) {
  const long long total  = (long long)n * h;
  const long long stride = (long long)gridDim.x * blockDim.x;
  for (long long i = (long long)blockIdx.x * blockDim.x + threadIdx.x;
       i < total; i += stride)
    out[i] = b[(int)(i % h)];
}

// ---------------------------------------------------------------------------
// GCN aggregation: agg[dst[e]][c] += h[src[e]][c] * ew[e]
// ---------------------------------------------------------------------------
__global__ void scatter_add(const float* __restrict__ h,
                            const int* __restrict__ src,
                            const int* __restrict__ dst,
                            const float* __restrict__ ew,
                            float* __restrict__ agg, int E, int H) {
  const long long total  = (long long)E * H;
  const long long stride = (long long)gridDim.x * blockDim.x;
  for (long long i = (long long)blockIdx.x * blockDim.x + threadIdx.x;
       i < total; i += stride) {
    const int e = (int)(i / H);
    const int c = (int)(i % H);
    atomicAdd(&agg[(long long)dst[e] * H + c],
              h[(long long)src[e] * H + c] * ew[e]);
  }
}

// ---------------------------------------------------------------------------
// Per-channel mean/var over N rows (population variance, matches jnp.var)
// ---------------------------------------------------------------------------
__global__ void bn_stats(const float* __restrict__ x, float* __restrict__ mean,
                         float* __restrict__ var, int n, int h) {
  const int c = blockIdx.x;
  float s = 0.f, s2 = 0.f;
  for (int i = threadIdx.x; i < n; i += blockDim.x) {
    const float v = x[(long long)i * h + c];
    s += v;
    s2 += v * v;
  }
  __shared__ float ss[256], ss2[256];
  ss[threadIdx.x]  = s;
  ss2[threadIdx.x] = s2;
  __syncthreads();
  for (int ofs = blockDim.x >> 1; ofs > 0; ofs >>= 1) {
    if ((int)threadIdx.x < ofs) {
      ss[threadIdx.x]  += ss[threadIdx.x + ofs];
      ss2[threadIdx.x] += ss2[threadIdx.x + ofs];
    }
    __syncthreads();
  }
  if (threadIdx.x == 0) {
    const float m = ss[0] / (float)n;
    mean[c] = m;
    var[c]  = ss2[0] / (float)n - m * m;
  }
}

__global__ void bn_relu(const float* __restrict__ x,
                        const float* __restrict__ mean,
                        const float* __restrict__ var,
                        const float* __restrict__ g,
                        const float* __restrict__ be,
                        float* __restrict__ y, int n, int h) {
  const long long total  = (long long)n * h;
  const long long stride = (long long)gridDim.x * blockDim.x;
  for (long long i = (long long)blockIdx.x * blockDim.x + threadIdx.x;
       i < total; i += stride) {
    const int c = (int)(i % h);
    const float v = (x[i] - mean[c]) * rsqrtf(var[c] + GEPS) * g[c] + be[c];
    y[i] = v > 0.f ? v : 0.f;
  }
}

// ---------------------------------------------------------------------------
// Sequential bi-LSTM recurrence. G = precomputed x@Wih^T + bih + bhh.
// Block 0: forward, block 1: backward. 512 threads: thread j owns gate row j
// of Whh. Whh (256 KB) lives ENTIRELY in LDS (CDNA5 WGP has 320 KB) with a
// padded row stride of 129 floats so the 32 lanes of a wave hit 32 distinct
// banks. The latency-critical inner loop is pure ds_load + v_fma.
// ---------------------------------------------------------------------------
__global__ void lstm_seq(const float* __restrict__ Gf,
                         const float* __restrict__ Gb,
                         const float* __restrict__ Whh_f,
                         const float* __restrict__ Whh_b,
                         float* __restrict__ hf, float* __restrict__ hb) {
  const bool fwd     = (blockIdx.x == 0);
  const float* G     = fwd ? Gf : Gb;
  const float* Whh   = fwd ? Whh_f : Whh_b;
  float* out         = fwd ? hf : hb;

  __shared__ float whh_s[G4H * GHP];  // 512*129*4 = 264,192 B  (<320 KB/WGP)
  __shared__ float h_s[GH];
  __shared__ float gates_s[G4H];
  const int j = threadIdx.x;  // 0..511

  // Stage Whh into LDS (coalesced global reads, once per call)
  for (int i = j; i < G4H * GH; i += G4H) {
    const int r = i >> 7;         // i / GH
    const int k = i & (GH - 1);   // i % GH
    whh_s[r * GHP + k] = Whh[i];
  }
  float c = 0.f;
  if (j < GH) h_s[j] = 0.f;
  __syncthreads();

  const float* wrow = &whh_s[j * GHP];
  for (int step = 0; step < GN; ++step) {
    const int t = fwd ? step : (GN - 1 - step);
    float gv = G[(long long)t * G4H + j];
#pragma unroll 8
    for (int k = 0; k < GH; ++k) gv = fmaf(h_s[k], wrow[k], gv);
    gates_s[j] = gv;
    __syncthreads();
    if (j < GH) {
      const float ig = sigf(gates_s[j]);
      const float fg = sigf(gates_s[GH + j]);
      const float gg = tanhf(gates_s[2 * GH + j]);
      const float og = sigf(gates_s[3 * GH + j]);
      c = fg * c + ig * gg;
      const float hn = og * tanhf(c);
      out[(long long)t * GH + j] = hn;
      h_s[j] = hn;
    }
    __syncthreads();
  }
}

// ---------------------------------------------------------------------------
// Attention is identity (softmax over length-1 axis) => out = [hf|hb]@Wo + bo
// ---------------------------------------------------------------------------
__global__ void out_proj(const float* __restrict__ hf,
                         const float* __restrict__ hb,
                         const float* __restrict__ Wo,
                         const float* __restrict__ bo,
                         float* __restrict__ out) {
  const long long total  = (long long)GN * GP;
  const long long stride = (long long)gridDim.x * blockDim.x;
  for (long long idx = (long long)blockIdx.x * blockDim.x + threadIdx.x;
       idx < total; idx += stride) {
    const int i = (int)(idx / GP);
    const int p = (int)(idx % GP);
    float s = bo[p];
    const float* hfr = hf + (long long)i * GH;
    const float* hbr = hb + (long long)i * GH;
#pragma unroll 4
    for (int k = 0; k < GH; ++k) {
      s = fmaf(hfr[k], Wo[k * GP + p], s);
      s = fmaf(hbr[k], Wo[(GH + k) * GP + p], s);
    }
    out[idx] = s;
  }
}

// ---------------------------------------------------------------------------
extern "C" void kernel_launch(void* const* d_in, const int* in_sizes, int n_in,
                              void* d_out, int out_size, void* d_ws,
                              size_t ws_size, hipStream_t stream) {
  const float* x     = (const float*)d_in[0];
  const int*   ei    = (const int*)d_in[1];
  const int*   src   = ei;
  const int*   dst   = ei + GE;
  const float* ew    = (const float*)d_in[2];
  const float* W1    = (const float*)d_in[3];
  const float* b1    = (const float*)d_in[4];
  const float* g1    = (const float*)d_in[5];
  const float* be1   = (const float*)d_in[6];
  const float* W2    = (const float*)d_in[7];
  const float* b2    = (const float*)d_in[8];
  const float* g2    = (const float*)d_in[9];
  const float* be2   = (const float*)d_in[10];
  const float* Wih_f = (const float*)d_in[11];
  const float* Whh_f = (const float*)d_in[12];
  const float* bih_f = (const float*)d_in[13];
  const float* bhh_f = (const float*)d_in[14];
  const float* Wih_b = (const float*)d_in[15];
  const float* Whh_b = (const float*)d_in[16];
  const float* bih_b = (const float*)d_in[17];
  const float* bhh_b = (const float*)d_in[18];
  // d_in[19] (Wa) / d_in[20] (ba): dead — softmax over length-1 axis is identity
  const float* Wo    = (const float*)d_in[21];
  const float* bo    = (const float*)d_in[22];
  float* out = (float*)d_out;

  // Workspace layout (floats): ~66.7 MB total, fits comfortably in L2 (192MB)
  float* ws   = (float*)d_ws;
  float* t0   = ws;                               // N*H   : gemm output
  float* agg  = t0   + (long long)GN * GH;        // N*H   : scatter accumulator
  float* hcur = agg  + (long long)GN * GH;        // N*H   : post-BN activations
  float* Gf   = hcur + (long long)GN * GH;        // N*4H  : fwd gate pre-acts
  float* Gb   = Gf   + (long long)GN * G4H;       // N*4H  : bwd gate pre-acts
  float* hfb  = Gb   + (long long)GN * G4H;       // N*H   : forward hidden seq
  float* hbb  = hfb  + (long long)GN * GH;        // N*H   : backward hidden seq
  float* mean = hbb  + (long long)GN * GH;        // H
  float* var  = mean + GH;                        // H
  float* W1T  = var  + GH;                        // H*IN_C : W1^T (128x64)
  float* W2T  = W1T  + GH * GINC;                 // H*H    : W2^T (128x128)

  const dim3 gblk(128);
  const dim3 ggrid_h(GH / 64, (GN + 63) / 64);    // Nc=128
  const dim3 ggrid_4h(G4H / 64, (GN + 63) / 64);  // Nc=512

  // ---- Weight transposes (tiny; makes every GEMM B-operand contiguous) ----
  transpose_mat<<<16, 256, 0, stream>>>(W1, W1T, GINC, GH);
  transpose_mat<<<32, 256, 0, stream>>>(W2, W2T, GH, GH);

  // ---- GCN layer 1 ----
  gemm_wmma_bf16<<<ggrid_h, gblk, 0, stream>>>(x, W1T, t0, GN, GINC, GH,
                                               nullptr, nullptr);
  fill_bias<<<1024, 256, 0, stream>>>(agg, b1, GN, GH);
  scatter_add<<<4096, 256, 0, stream>>>(t0, src, dst, ew, agg, GE, GH);
  bn_stats<<<GH, 256, 0, stream>>>(agg, mean, var, GN, GH);
  bn_relu<<<1024, 256, 0, stream>>>(agg, mean, var, g1, be1, hcur, GN, GH);

  // ---- GCN layer 2 ----
  gemm_wmma_bf16<<<ggrid_h, gblk, 0, stream>>>(hcur, W2T, t0, GN, GH, GH,
                                               nullptr, nullptr);
  fill_bias<<<1024, 256, 0, stream>>>(agg, b2, GN, GH);
  scatter_add<<<4096, 256, 0, stream>>>(t0, src, dst, ew, agg, GE, GH);
  bn_stats<<<GH, 256, 0, stream>>>(agg, mean, var, GN, GH);
  bn_relu<<<1024, 256, 0, stream>>>(agg, mean, var, g2, be2, hcur, GN, GH);

  // ---- LSTM input projections (the parallel part, hoisted out of the scan)
  //      G = h @ Wih^T + (bih + bhh); Wih is (4H,H) = [Nc,K] already.
  gemm_wmma_bf16<<<ggrid_4h, gblk, 0, stream>>>(hcur, Wih_f, Gf, GN, GH,
                                                G4H, bih_f, bhh_f);
  gemm_wmma_bf16<<<ggrid_4h, gblk, 0, stream>>>(hcur, Wih_b, Gb, GN, GH,
                                                G4H, bih_b, bhh_b);

  // ---- Sequential recurrences (latency-bound): 1 resident WG per direction,
  //      Whh LDS-resident (320 KB WGP LDS is the CDNA5 feature here)
  lstm_seq<<<2, G4H, 0, stream>>>(Gf, Gb, Whh_f, Whh_b, hfb, hbb);

  // ---- Output projection (attention is identity) ----
  out_proj<<<512, 256, 0, stream>>>(hfb, hbb, Wo, bo, out);
}
